// CGCNNRegressor_56985626084093
// MI455X (gfx1250) — compile-verified, matrix operand order checked by reading, not run on hardware
//
#include <hip/hip_runtime.h>
#include <hip/hip_bf16.h>
#include <cmath>

typedef __attribute__((ext_vector_type(16))) _Float16 v16h;
typedef __attribute__((ext_vector_type(8)))  _Float16 v8h;
typedef __attribute__((ext_vector_type(8)))  float    v8f;
typedef __attribute__((ext_vector_type(4)))  int      v4i;

typedef __attribute__((address_space(1))) v4i gv4i;   // global int4
typedef __attribute__((address_space(3))) v4i lv4i;   // LDS int4

#define NN 50000
#define EE 800000
#define GG 256
#define DD 64
#define ZD 178
#define KP 192          // Z padded to 6 K-chunks of 32
#define EDIM 50
#define LL 4
#define HH1 128
#define HH2 64
#define TILE 128        // edges per block
#define WPER 12288      // swizzled f16 weights per gate per layer (4*6*32*16)

#if __has_builtin(__builtin_amdgcn_global_load_async_to_lds_b128) && \
    __has_builtin(__builtin_amdgcn_s_wait_asynccnt)
#define USE_ASYNC_LDS 1
#else
#define USE_ASYNC_LDS 0
#endif

union AFrag { v16h v; v8h h[2]; };

// ---------------- embedding gather: h = emb[x], plus f16 shadow ----------------
__global__ void k_embed(const int* __restrict__ x, const float* __restrict__ emb,
                        float* __restrict__ h, _Float16* __restrict__ hh, int total) {
    int i = blockIdx.x * blockDim.x + threadIdx.x;
    if (i >= total) return;
    int n = i >> 6, d = i & 63;
    float v = emb[x[n] * DD + d];
    h[i] = v;
    hh[i] = (_Float16)v;
}

__global__ void k_zero(float* __restrict__ p, int total) {
    int i = blockIdx.x * blockDim.x + threadIdx.x;
    if (i < total) p[i] = 0.0f;
}

// ---------------- one-time weight prep: f32 -> f16, swizzled to B-fragment layout ----
// out layout: [L][gate][ ((nt*6+c)*32+lane)*16 + j ]
// lane<16 holds col N=nt*16+lane with K {0..7,16..23}; lane>=16: N=..(lane-16), K {8..15,24..31}
__global__ void k_prep(const float* __restrict__ Wf, const float* __restrict__ Ws,
                       _Float16* __restrict__ out, int total) {
    int t = blockIdx.x * blockDim.x + threadIdx.x;
    if (t >= total) return;
    int l    = t / (2 * WPER);
    int r    = t % (2 * WPER);
    int gate = r / WPER;
    int i    = r % WPER;
    int j    = i & 15;
    int lane = (i >> 4) & 31;
    int cn   = i >> 9;               // nt*6 + c
    int c    = cn % 6, nt = cn / 6;
    int K = c * 32 + ((lane >> 4) << 3) + (j < 8 ? j : j + 8);
    int n = nt * 16 + (lane & 15);
    const float* W = gate ? Ws : Wf;
    float v = (K < ZD) ? W[(size_t)l * ZD * DD + K * DD + n] : 0.0f;
    out[t] = (_Float16)v;
}

__device__ __forceinline__ void cp16_async(const void* g, void* l) {
#if USE_ASYNC_LDS
    __builtin_amdgcn_global_load_async_to_lds_b128(
        (gv4i*)g, (lv4i*)l, 0, 0);
#else
    *(uint4*)l = *(const uint4*)g;
#endif
}

// ---------------- CGConv edge kernel: WMMA gated message + scatter ----------------
// 256 threads = 8 waves; 128 edges per block; each wave computes 16 edges x 64 cols x 2 gates.
// B fragments read directly from pre-swizzled global weights (L2-resident, coalesced 32B/lane).
__global__ __launch_bounds__(256) void k_edge(
    const _Float16* __restrict__ hh, const float* __restrict__ eattr,
    const int* __restrict__ src, const int* __restrict__ dst,
    const _Float16* __restrict__ wfg, const _Float16* __restrict__ wsg,
    const float* __restrict__ bfv_g, const float* __restrict__ bsv_g,
    float* __restrict__ agg)
{
    __shared__ _Float16 z[TILE * KP];   // 48 KB: staged z rows (f16, padded)
    __shared__ int sdst[TILE];

    const int tid   = threadIdx.x;
    const int ebase = blockIdx.x * TILE;

    // ---- stage z rows: [h[dst] | h[src] | f16(edge_attr) | 0-pad] ----
    {
        int e = tid & (TILE - 1);
        int part = tid >> 7;
        int ge = ebase + e;
        if (part == 0) {
            int dn = dst[ge];
            sdst[e] = dn;
            const char* g = (const char*)(hh + (size_t)dn * DD);
            char* l = (char*)(&z[e * KP]);
            #pragma unroll
            for (int q = 0; q < 8; ++q) cp16_async(g + q * 16, l + q * 16);
        } else {
            int sn = src[ge];
            const char* g = (const char*)(hh + (size_t)sn * DD);
            char* l = (char*)(&z[e * KP + 64]);
            #pragma unroll
            for (int q = 0; q < 8; ++q) cp16_async(g + q * 16, l + q * 16);
            const float* ea = eattr + (size_t)ge * EDIM;
            #pragma unroll
            for (int d = 0; d < EDIM; ++d) z[e * KP + 128 + d] = (_Float16)ea[d];
            #pragma unroll
            for (int d = EDIM; d < 64; ++d) z[e * KP + 128 + d] = (_Float16)0.0f;
        }
    }
#if USE_ASYNC_LDS
    __builtin_amdgcn_s_wait_asynccnt(0);
#endif
    __syncthreads();

    const int lane = tid & 31;
    const int wave = tid >> 5;                  // 0..7, each owns 16 edge rows
    const int row  = wave * 16 + (lane & 15);   // edge row this lane carries in A
    const int koff = (lane >> 4) << 3;          // 0 or 8 per ISA A-fragment layout

    const v16h* wfF = (const v16h*)wfg;         // 768 fragments of 16 halves per gate
    const v16h* wsF = (const v16h*)wsg;

    v8f accf[4] = {};
    v8f accs[4] = {};

    #pragma unroll
    for (int c = 0; c < 6; ++c) {
        AFrag a;
        const _Float16* zp = &z[row * KP + c * 32 + koff];
        a.h[0] = *(const v8h*)(zp);        // K = base+koff .. +7
        a.h[1] = *(const v8h*)(zp + 16);   // K = base+koff+16 .. +23
        #pragma unroll
        for (int nt = 0; nt < 4; ++nt) {
            int fidx = (nt * 6 + c) * 32 + lane;
            v16h bf_frag = wfF[fidx];
            v16h bs_frag = wsF[fidx];
            accf[nt] = __builtin_amdgcn_wmma_f32_16x16x32_f16(
                false, a.v, false, bf_frag, (short)0, accf[nt], false, false);
            accs[nt] = __builtin_amdgcn_wmma_f32_16x16x32_f16(
                false, a.v, false, bs_frag, (short)0, accs[nt], false, false);
        }
    }

    // ---- epilogue: bias, sigmoid(f)*softplus(s), scatter-add over dst ----
    #pragma unroll
    for (int nt = 0; nt < 4; ++nt) {
        int col = nt * 16 + (lane & 15);
        float bfv = bfv_g[col];
        float bsv = bsv_g[col];
        #pragma unroll
        for (int v = 0; v < 8; ++v) {
            int M = v + ((lane >> 4) << 3);     // C layout: VGPR v -> row v / v+8
            int dn = sdst[wave * 16 + M];
            float fx = accf[nt][v] + bfv;
            float sx = accs[nt][v] + bsv;
            float gate = 1.0f / (1.0f + __expf(-fx));
            float sp   = fmaxf(sx, 0.0f) + log1pf(__expf(-fabsf(sx)));
            float m = gate * sp;
            __hip_atomic_fetch_add(&agg[(size_t)dn * DD + col], m,
                                   __ATOMIC_RELAXED, __HIP_MEMORY_SCOPE_AGENT);
        }
    }
}

// ---------------- residual + softplus; refresh f16 shadow ----------------
__global__ void k_update(float* __restrict__ h, const float* __restrict__ agg,
                         _Float16* __restrict__ hh, int total) {
    int i = blockIdx.x * blockDim.x + threadIdx.x;
    if (i >= total) return;
    float xv = h[i] + agg[i];
    float sp = fmaxf(xv, 0.0f) + log1pf(__expf(-fabsf(xv)));
    h[i] = sp;
    hh[i] = (_Float16)sp;
}

// ---------------- global mean pool (atomic sum + count) ----------------
__global__ void k_pool(const float* __restrict__ h, const int* __restrict__ batch,
                       float* __restrict__ pooled, float* __restrict__ cnt, int total) {
    int i = blockIdx.x * blockDim.x + threadIdx.x;
    if (i >= total) return;
    int n = i >> 6, d = i & 63;
    int g = batch[n];
    __hip_atomic_fetch_add(&pooled[(size_t)g * DD + d], h[i],
                           __ATOMIC_RELAXED, __HIP_MEMORY_SCOPE_AGENT);
    if (d == 0)
        __hip_atomic_fetch_add(&cnt[g], 1.0f,
                               __ATOMIC_RELAXED, __HIP_MEMORY_SCOPE_AGENT);
}

// ---------------- tiny readout MLP: one thread per graph ----------------
__global__ __launch_bounds__(256) void k_mlp(
    const float* __restrict__ pooled, const float* __restrict__ cnt,
    const float* __restrict__ W1, const float* __restrict__ b1,
    const float* __restrict__ W2, const float* __restrict__ b2,
    const float* __restrict__ Wo, const float* __restrict__ bo,
    float* __restrict__ out)
{
    int g = threadIdx.x;   // grid = 1 block of 256
    float p[DD];
    float inv = 1.0f / fmaxf(cnt[g], 1.0f);
    for (int d = 0; d < DD; ++d) p[d] = pooled[(size_t)g * DD + d] * inv;

    float g1[HH1];
    for (int k = 0; k < HH1; ++k) {
        float acc = b1[k];
        for (int d = 0; d < DD; ++d) acc += p[d] * W1[d * HH1 + k];
        g1[k] = fmaxf(acc, 0.0f);
    }
    float g2[HH2];
    for (int j = 0; j < HH2; ++j) {
        float acc = b2[j];
        for (int k = 0; k < HH1; ++k) acc += g1[k] * W2[k * HH2 + j];
        g2[j] = fmaxf(acc, 0.0f);
    }
    float o = bo[0];
    for (int j = 0; j < HH2; ++j) o += g2[j] * Wo[j];
    out[g] = o;
}

extern "C" void kernel_launch(void* const* d_in, const int* in_sizes, int n_in,
                              void* d_out, int out_size, void* d_ws, size_t ws_size,
                              hipStream_t stream) {
    const int*   x     = (const int*)  d_in[0];
    const int*   ei    = (const int*)  d_in[1];
    const float* eattr = (const float*)d_in[2];
    const int*   batch = (const int*)  d_in[3];
    const float* emb   = (const float*)d_in[4];
    const float* Wf    = (const float*)d_in[5];
    const float* bf    = (const float*)d_in[6];
    const float* Ws    = (const float*)d_in[7];
    const float* bs    = (const float*)d_in[8];
    const float* W1    = (const float*)d_in[9];
    const float* b1    = (const float*)d_in[10];
    const float* W2    = (const float*)d_in[11];
    const float* b2    = (const float*)d_in[12];
    const float* Wo    = (const float*)d_in[13];
    const float* bo    = (const float*)d_in[14];
    float* out = (float*)d_out;

    char* ws = (char*)d_ws;
    size_t off = 0;
    float*     h      = (float*)(ws + off); off += (size_t)NN * DD * 4;      // 12.8 MB
    _Float16*  hh     = (_Float16*)(ws + off); off += (size_t)NN * DD * 2;   // 6.4 MB
    float*     agg    = (float*)(ws + off); off += (size_t)NN * DD * 4;      // 12.8 MB
    float*     pooled = (float*)(ws + off); off += (size_t)GG * DD * 4;      // 64 KB
    float*     cnt    = (float*)(ws + off); off += (size_t)GG * 4;           // 1 KB
    off = (off + 255) & ~(size_t)255;
    _Float16*  wswz   = (_Float16*)(ws + off); off += (size_t)LL * 2 * WPER * 2; // 192 KB

    const int* srcp = ei;        // edge_index[0]
    const int* dstp = ei + EE;   // edge_index[1]

    const int totalND = NN * DD;
    const int totalW  = LL * 2 * WPER;
    k_prep<<<(totalW + 255) / 256, 256, 0, stream>>>(Wf, Ws, wswz, totalW);
    k_embed<<<(totalND + 255) / 256, 256, 0, stream>>>(x, emb, h, hh, totalND);

    for (int l = 0; l < LL; ++l) {
        k_zero<<<(totalND + 255) / 256, 256, 0, stream>>>(agg, totalND);
        k_edge<<<EE / TILE, 256, 0, stream>>>(hh, eattr, srcp, dstp,
                                              wswz + (size_t)(l * 2 + 0) * WPER,
                                              wswz + (size_t)(l * 2 + 1) * WPER,
                                              bf + (size_t)l * DD,
                                              bs + (size_t)l * DD,
                                              agg);
        k_update<<<(totalND + 255) / 256, 256, 0, stream>>>(h, agg, hh, totalND);
    }

    const int totalP = GG * DD + GG;   // pooled and cnt are contiguous
    k_zero<<<(totalP + 255) / 256, 256, 0, stream>>>(pooled, totalP);
    k_pool<<<(totalND + 255) / 256, 256, 0, stream>>>(h, batch, pooled, cnt, totalND);
    k_mlp<<<1, 256, 0, stream>>>(pooled, cnt, W1, b1, W2, b2, Wo, bo, out);
}
